// GAT_12421045420832
// MI455X (gfx1250) — matrix operand order repaired and checked by reading, not compiled, not run on hardware
//
#include <hip/hip_runtime.h>
#include <cstdint>
#include <cstddef>

typedef unsigned short u16;
typedef __attribute__((ext_vector_type(8)))  u16    v8u;
typedef __attribute__((ext_vector_type(16))) u16    v16u;
typedef __attribute__((ext_vector_type(16))) __bf16 v16bf;
typedef __attribute__((ext_vector_type(8)))  float  v8f;

#define NEG_SLOPE 0.2f

__device__ __forceinline__ u16 f2bf(float x) {
  unsigned u = __float_as_uint(x);
  unsigned r = u + 0x7FFFu + ((u >> 16) & 1u);   // round-to-nearest-even
  return (u16)(r >> 16);
}
__device__ __forceinline__ unsigned encf(float x) {  // order-preserving f32->u32
  unsigned u = __float_as_uint(x);
  return (u & 0x80000000u) ? ~u : (u | 0x80000000u);
}
__device__ __forceinline__ float decf(unsigned k) {
  unsigned u = (k & 0x80000000u) ? (k & 0x7FFFFFFFu) : ~k;
  return __uint_as_float(u);
}
__device__ __forceinline__ float lrelu(float v) { return v > 0.f ? v : NEG_SLOPE * v; }

// ---------------- elementwise prep ----------------
__global__ void k_cast_bf16(const float* __restrict__ src, u16* __restrict__ dst, int n) {
  int i = blockIdx.x * blockDim.x + threadIdx.x;
  if (i < n) dst[i] = f2bf(src[i]);
}

// W [K, Nsrc] f32 row-major -> Bt [Npad, K] bf16 (rows >= Nsrc zero-filled)
__global__ void k_transpose_bf16(const float* __restrict__ W, u16* __restrict__ Bt,
                                 int K, int Nsrc, int Npad) {
  int i = blockIdx.x * blockDim.x + threadIdx.x;
  if (i >= Npad * K) return;
  int n = i / K, k = i - n * K;
  Bt[i] = (n < Nsrc) ? f2bf(W[(size_t)k * Nsrc + n]) : (u16)0;
}

// ---------------- WMMA GEMM: C[M,N] f32 = A[M,K]bf16 @ Bt[N,K]bf16^T ----------------
// block = (32,4): 4 waves, each computes a 16(M) x 64(N) tile. The K-loop is
// software-pipelined (register double buffering): iteration k+1's loads issue
// before iteration k's wmmas, so the scheduler can emit partial s_wait_loadcnt
// and overlap L2 latency with matrix issue.
// Requirements: M % 64 == 0, K % 32 == 0, N % 64 == 0.
__device__ __forceinline__ v16bf load_afrag(const u16* __restrict__ arow, int k0, int g) {
  // 16-bit A 16x32 layout: elem e -> K=k0+16g+e (contiguous, matches B)
  v8u alo = *(const v8u*)(arow + k0 + 16 * g);
  v8u ahi = *(const v8u*)(arow + k0 + 16 * g + 8);
  v16u av;
#pragma unroll
  for (int i = 0; i < 8; ++i) { av[i] = alo[i]; av[i + 8] = ahi[i]; }
  return __builtin_bit_cast(v16bf, av);
}
__device__ __forceinline__ v16bf load_bfrag(const u16* __restrict__ b, int k0) {
  // 16-bit B 32x16 layout: N = lane%16, elem e -> K = k0+16g+e (contiguous in Bt)
  return __builtin_bit_cast(v16bf, *(const v16u*)(b + k0));
}

__global__ __launch_bounds__(128)
void k_gemm_bf16(const u16* __restrict__ A, const u16* __restrict__ Bt,
                 float* __restrict__ C, int M, int K, int N) {
  const int lane = threadIdx.x;
  const int wave = threadIdx.y;
  const int m0 = (blockIdx.y * 4 + wave) * 16;
  const int n0 = blockIdx.x * 64;
  const int g = lane >> 4;                // lane group (0/1)
  const int r = lane & 15;

  v8f acc[4];
#pragma unroll
  for (int s = 0; s < 4; ++s)
#pragma unroll
    for (int i = 0; i < 8; ++i) acc[s][i] = 0.f;

  const u16* arow = A + (size_t)(m0 + r) * K;       // A fragment row: M = lane%16
  const u16* brow = Bt + (size_t)(n0 + r) * K + 16 * g;

  // prologue: fragments for k0 = 0
  v16bf a_cur = load_afrag(arow, 0, g);
  v16bf b_cur[4];
#pragma unroll
  for (int s = 0; s < 4; ++s) b_cur[s] = load_bfrag(brow + (size_t)(16 * s) * K, 0);

#pragma unroll 2
  for (int k0 = 32; k0 < K; k0 += 32) {
    // issue next iteration's loads first
    v16bf a_nxt = load_afrag(arow, k0, g);
    v16bf b_nxt[4];
#pragma unroll
    for (int s = 0; s < 4; ++s) b_nxt[s] = load_bfrag(brow + (size_t)(16 * s) * K, k0);
    // consume current fragments
    acc[0] = __builtin_amdgcn_wmma_f32_16x16x32_bf16(false, a_cur, false, b_cur[0], (short)0, acc[0], false, false);
    acc[1] = __builtin_amdgcn_wmma_f32_16x16x32_bf16(false, a_cur, false, b_cur[1], (short)0, acc[1], false, false);
    acc[2] = __builtin_amdgcn_wmma_f32_16x16x32_bf16(false, a_cur, false, b_cur[2], (short)0, acc[2], false, false);
    acc[3] = __builtin_amdgcn_wmma_f32_16x16x32_bf16(false, a_cur, false, b_cur[3], (short)0, acc[3], false, false);
    a_cur = a_nxt;
#pragma unroll
    for (int s = 0; s < 4; ++s) b_cur[s] = b_nxt[s];
  }
  // epilogue
  acc[0] = __builtin_amdgcn_wmma_f32_16x16x32_bf16(false, a_cur, false, b_cur[0], (short)0, acc[0], false, false);
  acc[1] = __builtin_amdgcn_wmma_f32_16x16x32_bf16(false, a_cur, false, b_cur[1], (short)0, acc[1], false, false);
  acc[2] = __builtin_amdgcn_wmma_f32_16x16x32_bf16(false, a_cur, false, b_cur[2], (short)0, acc[2], false, false);
  acc[3] = __builtin_amdgcn_wmma_f32_16x16x32_bf16(false, a_cur, false, b_cur[3], (short)0, acc[3], false, false);

#pragma unroll
  for (int s = 0; s < 4; ++s) {
    // f32 C/D layout: VGPR i -> M = m0 + 8g + i, N = n0 + s*16 + lane%16
    float* crow = C + (size_t)(m0 + 8 * g) * N + (n0 + s * 16 + r);
#pragma unroll
    for (int i = 0; i < 8; ++i) crow[(size_t)i * N] = acc[s][i];
  }
}

// ---------------- attention halves: el/er[row] = sum_d z[row,d]*a[(g%4)*D+d] ----------------
__global__ __launch_bounds__(128)
void k_attn(const float* __restrict__ z, const float* __restrict__ al,
            const float* __restrict__ ar, float* __restrict__ el, float* __restrict__ er,
            int rows, int HP, int D) {
  int row = blockIdx.x * 4 + threadIdx.y;
  if (row >= rows) return;
  int lane = threadIdx.x;
  int g = row % HP;
  const float* zr  = z  + (size_t)row * D;
  const float* alv = al + (size_t)(g & 3) * D;   // heads = 4
  const float* arv = ar + (size_t)(g & 3) * D;
  float sl = 0.f, sr = 0.f;
  for (int d = lane; d < D; d += 32) {
    float zv = zr[d];
    sl += zv * alv[d];
    sr += zv * arv[d];
  }
  for (int m = 16; m > 0; m >>= 1) { sl += __shfl_xor(sl, m, 32); sr += __shfl_xor(sr, m, 32); }
  if (lane == 0) { el[row] = sl; er[row] = sr; }
}

__global__ void k_init_attn(unsigned* __restrict__ emax, float* __restrict__ denom, int n) {
  int i = blockIdx.x * blockDim.x + threadIdx.x;
  if (i < n) { emax[i] = 0x007FFFFFu /* enc(-inf) */; denom[i] = 0.f; }
}

__global__ void k_edge_max(const int* __restrict__ src, const int* __restrict__ dst,
                           const float* __restrict__ el, const float* __restrict__ er,
                           unsigned* __restrict__ emax, int E, int HP) {
  int i = blockIdx.x * blockDim.x + threadIdx.x;
  if (i >= E * HP) return;
  int e = i / HP, g = i - e * HP;
  int s = src[e], d = dst[e];
  float v = lrelu(el[s * HP + g] + er[d * HP + g]);
  atomicMax(&emax[d * HP + g], encf(v));
}

__global__ void k_edge_expsum(const int* __restrict__ src, const int* __restrict__ dst,
                              const float* __restrict__ el, const float* __restrict__ er,
                              const unsigned* __restrict__ emax, float* __restrict__ denom,
                              int E, int HP) {
  int i = blockIdx.x * blockDim.x + threadIdx.x;
  if (i >= E * HP) return;
  int e = i / HP, g = i - e * HP;
  int s = src[e], d = dst[e];
  float v = lrelu(el[s * HP + g] + er[d * HP + g]);
  float ex = expf(v - decf(emax[d * HP + g]));
  atomicAdd(&denom[d * HP + g], ex);
}

// one 256-thread block per edge: acc[dst, g, :] += a_g * z[src, g, :]
__global__ __launch_bounds__(256)
void k_edge_agg(const int* __restrict__ src, const int* __restrict__ dst,
                const float* __restrict__ el, const float* __restrict__ er,
                const unsigned* __restrict__ emax, const float* __restrict__ denom,
                const float* __restrict__ z, float* __restrict__ acc,
                int HP, int dshift) {
  __shared__ float a_sh[16];
  int e = blockIdx.x;
  int tid = threadIdx.x;
  int s = src[e], d = dst[e];
  if (tid < HP) {
    float v = lrelu(el[s * HP + tid] + er[d * HP + tid]);
    float ex = expf(v - decf(emax[d * HP + tid]));
    float dn = denom[d * HP + tid];
    a_sh[tid] = ex / (dn == 0.f ? 1.f : dn);
  }
  __syncthreads();
  int row = HP << dshift;
  const float* zs = z + (size_t)s * row;
  float*       ad = acc + (size_t)d * row;
  for (int j = tid; j < row; j += 256)
    atomicAdd(&ad[j], a_sh[j >> dshift] * zs[j]);
}

__global__ void k_bias_relu_bf16(const float* __restrict__ acc, const float* __restrict__ bias,
                                 u16* __restrict__ outb, int total, int rowlen) {
  int i = blockIdx.x * blockDim.x + threadIdx.x;
  if (i >= total) return;
  float v = acc[i] + bias[i % rowlen];
  outb[i] = f2bf(v > 0.f ? v : 0.f);
}

__global__ void k_out_bias(const float* __restrict__ c3, const float* __restrict__ bout,
                           float* __restrict__ out, int total, int CLS, int ldc) {
  int i = blockIdx.x * blockDim.x + threadIdx.x;
  if (i >= total) return;
  int nrow = i / CLS, c = i - nrow * CLS;
  out[i] = c3[(size_t)nrow * ldc + c] + bout[c];
}

// ---------------- host ----------------
static inline int cdiv(int a, int b) { return (a + b - 1) / b; }

extern "C" void kernel_launch(void* const* d_in, const int* in_sizes, int n_in,
                              void* d_out, int out_size, void* d_ws, size_t ws_size,
                              hipStream_t stream) {
  (void)in_sizes; (void)n_in; (void)out_size; (void)ws_size;
  const float* features = (const float*)d_in[0];
  const int*   src      = (const int*)  d_in[1];
  const int*   dst      = (const int*)  d_in[2];
  const float* W1       = (const float*)d_in[3];
  const float* al1      = (const float*)d_in[4];
  const float* ar1      = (const float*)d_in[5];
  const float* resW1    = (const float*)d_in[6];
  const float* b1       = (const float*)d_in[7];
  const float* W2       = (const float*)d_in[8];
  const float* al2      = (const float*)d_in[9];
  const float* ar2      = (const float*)d_in[10];
  const float* resW2    = (const float*)d_in[11];
  const float* b2       = (const float*)d_in[12];
  const float* Wout     = (const float*)d_in[13];
  const float* bout     = (const float*)d_in[14];
  float* out = (float*)d_out;

  const int Nn = 8192, Ed = 65536, IN = 256, HID = 512, H = 4, O2 = 32;
  const int CLS = 40, CLSP = 64;          // head N padded to 64 -> no partial tiles
  const int R1 = H * HID;                 // 2048 per-node row, layer 1
  const int M2 = Nn * H;                  // 32768 GEMM rows, layer 2
  const int N2 = H * O2;                  // 128
  const int HP2 = 16, R2 = HP2 * O2;      // 512 per-node row, layer 2

  // ---- workspace carving ----
  char* base = (char*)d_ws;
  size_t off = 0;
  auto carve = [&](size_t bytes) { void* p = base + off; off = (off + bytes + 255) & ~(size_t)255; return p; };
  u16*   featb  = (u16*)  carve((size_t)Nn * IN * 2);
  u16*   W1t    = (u16*)  carve((size_t)R1 * IN * 2);
  u16*   rW1t   = (u16*)  carve((size_t)R1 * IN * 2);
  u16*   W2t    = (u16*)  carve((size_t)N2 * HID * 2);
  u16*   rW2t   = (u16*)  carve((size_t)N2 * HID * 2);
  u16*   Woutt  = (u16*)  carve((size_t)CLSP * HID * 2);
  float* z1     = (float*)carve((size_t)Nn * R1 * 4);   // 64MB region (reused below)
  float* acc1   = (float*)carve((size_t)Nn * R1 * 4);   // 64MB region (reused below)
  float* el1    = (float*)carve((size_t)Nn * H * 4);
  float* er1    = (float*)carve((size_t)Nn * H * 4);
  unsigned* emax1 = (unsigned*)carve((size_t)Nn * H * 4);
  float* den1   = (float*)carve((size_t)Nn * H * 4);
  u16*   h1b    = (u16*)  carve((size_t)Nn * R1 * 2);
  float* el2    = (float*)carve((size_t)Nn * HP2 * 4);
  float* er2    = (float*)carve((size_t)Nn * HP2 * 4);
  unsigned* emax2 = (unsigned*)carve((size_t)Nn * HP2 * 4);
  float* den2   = (float*)carve((size_t)Nn * HP2 * 4);
  // region reuse (layer-1 big buffers dead by the time these are written):
  float* z2   = z1;                                        // 16MB into 64MB region
  u16*   h2b  = (u16*)((char*)z1 + (size_t)32 * 1024 * 1024);
  float* acc2 = acc1;                                      // 16MB into 64MB region
  float* c3   = (float*)((char*)acc1 + (size_t)32 * 1024 * 1024);

  const int TB = 256;

  // ---- prep: casts & weight transposes ----
  k_cast_bf16<<<cdiv(Nn * IN, TB), TB, 0, stream>>>(features, featb, Nn * IN);
  k_transpose_bf16<<<cdiv(R1 * IN, TB), TB, 0, stream>>>(W1,    W1t,  IN,  R1,  R1);
  k_transpose_bf16<<<cdiv(R1 * IN, TB), TB, 0, stream>>>(resW1, rW1t, IN,  R1,  R1);
  k_transpose_bf16<<<cdiv(N2 * HID, TB), TB, 0, stream>>>(W2,    W2t,  HID, N2,  N2);
  k_transpose_bf16<<<cdiv(N2 * HID, TB), TB, 0, stream>>>(resW2, rW2t, HID, N2,  N2);
  k_transpose_bf16<<<cdiv(CLSP * HID, TB), TB, 0, stream>>>(Wout, Woutt, HID, CLS, CLSP);

  dim3 blkG(32, 4);
  // ---- layer 1 ----
  k_gemm_bf16<<<dim3(R1 / 64, Nn / 64), blkG, 0, stream>>>(featb, W1t,  z1,   Nn, IN, R1);
  k_gemm_bf16<<<dim3(R1 / 64, Nn / 64), blkG, 0, stream>>>(featb, rW1t, acc1, Nn, IN, R1);
  k_attn<<<dim3((Nn * H) / 4), dim3(32, 4), 0, stream>>>(z1, al1, ar1, el1, er1, Nn * H, H, HID);
  k_init_attn<<<cdiv(Nn * H, TB), TB, 0, stream>>>(emax1, den1, Nn * H);
  k_edge_max<<<cdiv(Ed * H, TB), TB, 0, stream>>>(src, dst, el1, er1, emax1, Ed, H);
  k_edge_expsum<<<cdiv(Ed * H, TB), TB, 0, stream>>>(src, dst, el1, er1, emax1, den1, Ed, H);
  k_edge_agg<<<dim3(Ed), dim3(256), 0, stream>>>(src, dst, el1, er1, emax1, den1, z1, acc1, H, 9);
  k_bias_relu_bf16<<<cdiv(Nn * R1, TB), TB, 0, stream>>>(acc1, b1, h1b, Nn * R1, R1);

  // ---- layer 2 (A = h1b viewed as [32768, 512]) ----
  k_gemm_bf16<<<dim3(N2 / 64, M2 / 64), blkG, 0, stream>>>(h1b, W2t,  z2,   M2, HID, N2);
  k_gemm_bf16<<<dim3(N2 / 64, M2 / 64), blkG, 0, stream>>>(h1b, rW2t, acc2, M2, HID, N2);
  k_attn<<<dim3((Nn * HP2) / 4), dim3(32, 4), 0, stream>>>(z2, al2, ar2, el2, er2, Nn * HP2, HP2, O2);
  k_init_attn<<<cdiv(Nn * HP2, TB), TB, 0, stream>>>(emax2, den2, Nn * HP2);
  k_edge_max<<<cdiv(Ed * HP2, TB), TB, 0, stream>>>(src, dst, el2, er2, emax2, Ed, HP2);
  k_edge_expsum<<<cdiv(Ed * HP2, TB), TB, 0, stream>>>(src, dst, el2, er2, emax2, den2, Ed, HP2);
  k_edge_agg<<<dim3(Ed), dim3(256), 0, stream>>>(src, dst, el2, er2, emax2, den2, z2, acc2, HP2, 5);
  k_bias_relu_bf16<<<cdiv(M2 * N2, TB), TB, 0, stream>>>(acc2, b2, h2b, M2 * N2, N2);

  // ---- head: [8192, 512] @ [512, 40] (padded to 64) ----
  k_gemm_bf16<<<dim3(CLSP / 64, Nn / 64), blkG, 0, stream>>>(h2b, Woutt, c3, Nn, R2, CLSP);
  k_out_bias<<<cdiv(Nn * CLS, TB), TB, 0, stream>>>(c3, bout, out, Nn * CLS, CLS, CLSP);
}
